// MultiHeadAttention_23733989278260
// MI455X (gfx1250) — compile-verified, matrix operand order checked by reading, not run on hardware
//
#include <hip/hip_runtime.h>
#include <hip/hip_bf16.h>

// ---------------------------------------------------------------------------
// MHA forward for T=2048, D=1024, H=16, Dh=64 on gfx1250 (CDNA5, wave32).
// All matmuls via V_WMMA_F32_16X16X32_BF16 (f32 accumulate).
// Round 2: double-buffered GEMM loop with distinct B-operand registers,
// V-loads overlapped with softmax, exp2-domain softmax.
// ---------------------------------------------------------------------------

#define T_SEQ  2048
#define DMODEL 1024
#define NHEAD  16
#define DHEAD  64

typedef __attribute__((ext_vector_type(16))) __bf16 v16bf;
typedef __attribute__((ext_vector_type(8)))  __bf16 v8bf;
typedef __attribute__((ext_vector_type(8)))  float  v8f;

union ABf { v16bf v; v8bf h[2]; };

static __device__ inline v8f wmma_bf16(const ABf& a, const ABf& b, v8f c) {
  // D = A(16x32 bf16) * B(32x16 bf16) + C(16x16 f32)
  return __builtin_amdgcn_wmma_f32_16x16x32_bf16(
      /*neg_a=*/false, a.v, /*neg_b=*/false, b.v,
      /*c_mod=*/(short)0, c, /*reuse_a=*/false, /*reuse_b=*/false);
}

// Reference RoPE degenerates to element-wise scale:
//   scale(t,j) = cos(t*theta) + sin(t*theta)  (j < 512)
//              = cos(t*theta) - sin(t*theta)  (j >= 512)
//   theta = 10000^{-(j mod 512)/512}
static __device__ inline float rope_scale(int t, int j) {
  int i = j & 511;
  float theta = __expf(-9.210340371976184f * (float)i * (1.0f / 512.0f));
  float ang = (float)t * theta;
  float s, c;
  __sincosf(ang, &s, &c);
  return (j < 512) ? (c + s) : (c - s);
}

// ------------------------------ prep kernels -------------------------------

__global__ void cvt_f32_bf16(const float* __restrict__ src,
                             __bf16* __restrict__ dst, int n) {
  int i = blockIdx.x * blockDim.x + threadIdx.x;
  if (i < n) dst[i] = (__bf16)src[i];
}

// W: [K][N] row-major fp32  ->  WT: [N][K] bf16 (K contiguous per row)
__global__ void transpose_bf16(const float* __restrict__ W,
                               __bf16* __restrict__ WT, int K, int N) {
  int k = blockIdx.x * blockDim.x + threadIdx.x;
  int n = blockIdx.y;
  if (k < K) WT[(size_t)n * K + k] = (__bf16)W[(size_t)k * N + n];
}

// ------------------------------ WMMA GEMM ----------------------------------
// C[T][1024] = A[T][1024](bf16) * BT[1024][1024]^T(bf16) + bias
// One wave computes a 16x64 tile, double-buffered over K chunks of 32.
// mode 0: out bf16 [T][1024], epilogue multiplies by rope_scale   (Q / K)
// mode 1: out bf16 transposed [1024][T]                            (V)
// mode 2: out f32  [T][1024]                                       (final)
__global__ void __launch_bounds__(256)
gemm_wmma(const __bf16* __restrict__ A, const __bf16* __restrict__ BT,
          const float* __restrict__ bias, void* __restrict__ out, int mode) {
  const int lane = threadIdx.x & 31;
  const int wave = threadIdx.x >> 5;
  const int w    = blockIdx.x * 8 + wave;   // 0..2047
  const int tt   = w & 127;                 // 128 row tiles of 16
  const int nn   = w >> 7;                  // 16 col tiles of 64
  const int t0   = tt * 16;
  const int n0   = nn * 64;
  const int half = lane >> 4;
  const int l16  = lane & 15;

  v8f acc[4];
#pragma unroll
  for (int n = 0; n < 4; ++n)
#pragma unroll
    for (int r = 0; r < 8; ++r) acc[n][r] = 0.0f;

  const __bf16* arow = A + (size_t)(t0 + l16) * DMODEL;
  const __bf16* brow[4];
#pragma unroll
  for (int n = 0; n < 4; ++n)
    brow[n] = BT + (size_t)(n0 + n * 16 + l16) * DMODEL;

  // ---- software pipeline: cur operands in distinct registers, preload next
  ABf a, b[4];
  {
    const int kb = half * 8;
    a.h[0] = *(const v8bf*)(arow + kb);
    a.h[1] = *(const v8bf*)(arow + kb + 16);
#pragma unroll
    for (int n = 0; n < 4; ++n) {
      b[n].h[0] = *(const v8bf*)(brow[n] + kb);
      b[n].h[1] = *(const v8bf*)(brow[n] + kb + 16);
    }
  }
  for (int kc = 0; kc < DMODEL / 32; ++kc) {
    const int kn = (kc + 1 < DMODEL / 32) ? kc + 1 : kc;  // clamp (redundant last)
    const int kb = kn * 32 + half * 8;
    ABf a2, b2[4];
    a2.h[0] = *(const v8bf*)(arow + kb);
    a2.h[1] = *(const v8bf*)(arow + kb + 16);
#pragma unroll
    for (int n = 0; n < 4; ++n) {
      b2[n].h[0] = *(const v8bf*)(brow[n] + kb);
      b2[n].h[1] = *(const v8bf*)(brow[n] + kb + 16);
    }
#pragma unroll
    for (int n = 0; n < 4; ++n) acc[n] = wmma_bf16(a, b[n], acc[n]);
    a = a2;
#pragma unroll
    for (int n = 0; n < 4; ++n) b[n] = b2[n];
  }

#pragma unroll
  for (int n = 0; n < 4; ++n) {
    const int col = n0 + n * 16 + l16;
    const float bv = bias[col];
#pragma unroll
    for (int r = 0; r < 8; ++r) {
      const int row = t0 + r + 8 * half;     // C layout: VGPR r, half selects M/M+8
      float v = acc[n][r] + bv;
      if (mode == 0) {
        v *= rope_scale(row, col);
        ((__bf16*)out)[(size_t)row * DMODEL + col] = (__bf16)v;
      } else if (mode == 1) {
        ((__bf16*)out)[(size_t)col * T_SEQ + row] = (__bf16)v;  // V transposed
      } else {
        ((float*)out)[(size_t)row * DMODEL + col] = v;
      }
    }
  }
}

// --------------------------- flash attention -------------------------------
// One wave owns (head h, 16 query rows). Streams keys in 32-wide blocks with
// online softmax (exp2 domain); P (16x32) bounces C->A layout through LDS.
// Q,K: bf16 [T][1024] (rope'd). Vt: bf16 [1024][T]. O: bf16 [T][1024].
__global__ void __launch_bounds__(256)
flash_attn(const __bf16* __restrict__ Q, const __bf16* __restrict__ K,
           const __bf16* __restrict__ Vt, __bf16* __restrict__ O) {
  __shared__ __bf16 lds[8 * 16 * 32];       // 1KB per wave
  const int lane = threadIdx.x & 31;
  const int wave = threadIdx.x >> 5;
  const int w    = blockIdx.x * 8 + wave;   // 0..2047
  const int h    = w >> 7;                  // head
  const int tt   = w & 127;                 // row tile
  const int t0   = tt * 16;
  const int half = lane >> 4;
  const int l16  = lane & 15;
  __bf16* pl = lds + wave * 512;

  // scores scaled into log2 domain: raw * (1/sqrt(64)) * log2(e)
  const float SCL = 0.125f * 1.4426950408889634f;

  // Q as two A operands (dh chunks 0..31 and 32..63)
  ABf qa[2];
  {
    const __bf16* qrow = Q + (size_t)(t0 + l16) * DMODEL + h * DHEAD;
#pragma unroll
    for (int c = 0; c < 2; ++c) {
      qa[c].h[0] = *(const v8bf*)(qrow + c * 32 + half * 8);
      qa[c].h[1] = *(const v8bf*)(qrow + c * 32 + 16 + half * 8);
    }
  }

  float m[8], l[8];
  v8f acc[4];
#pragma unroll
  for (int r = 0; r < 8; ++r) { m[r] = -3.0e38f; l[r] = 0.0f; }
#pragma unroll
  for (int n = 0; n < 4; ++n)
#pragma unroll
    for (int r = 0; r < 8; ++r) acc[n][r] = 0.0f;

  for (int jb = 0; jb < T_SEQ / 32; ++jb) {
    // ---- load all four K B-operands (distinct registers) ----
    ABf kb[4];  // [n*2+cc]
#pragma unroll
    for (int n = 0; n < 2; ++n) {
      const __bf16* krow =
          K + (size_t)(jb * 32 + n * 16 + l16) * DMODEL + h * DHEAD;
#pragma unroll
      for (int cc = 0; cc < 2; ++cc) {
        kb[n * 2 + cc].h[0] = *(const v8bf*)(krow + cc * 32 + half * 8);
        kb[n * 2 + cc].h[1] = *(const v8bf*)(krow + cc * 32 + 16 + half * 8);
      }
    }
    // ---- S = Q * K^T (two 16x16 tiles, back-to-back WMMAs) ----
    v8f s[2];
#pragma unroll
    for (int n = 0; n < 2; ++n) {
      v8f c;
#pragma unroll
      for (int r = 0; r < 8; ++r) c[r] = 0.0f;
      c = wmma_bf16(qa[0], kb[n * 2 + 0], c);
      c = wmma_bf16(qa[1], kb[n * 2 + 1], c);
      s[n] = c;
    }

    // ---- issue V loads now; they overlap the softmax VALU work ----
    ABf vb[4];
#pragma unroll
    for (int n = 0; n < 4; ++n) {
      const __bf16* vrow =
          Vt + (size_t)(h * DHEAD + n * 16 + l16) * T_SEQ + jb * 32;
      vb[n].h[0] = *(const v8bf*)(vrow + half * 8);
      vb[n].h[1] = *(const v8bf*)(vrow + 16 + half * 8);
    }

    // ---- online softmax in exp2 domain (rows r+8*half in 16-lane halves) --
#pragma unroll
    for (int r = 0; r < 8; ++r) {
      float s0 = s[0][r] * SCL;
      float s1 = s[1][r] * SCL;
      float mloc = fmaxf(s0, s1);
#pragma unroll
      for (int d = 1; d < 16; d <<= 1) mloc = fmaxf(mloc, __shfl_xor(mloc, d, 32));
      float mnew = fmaxf(m[r], mloc);
      float corr = __builtin_amdgcn_exp2f(m[r] - mnew);
      float p0 = __builtin_amdgcn_exp2f(s0 - mnew);
      float p1 = __builtin_amdgcn_exp2f(s1 - mnew);
      float ls = p0 + p1;
#pragma unroll
      for (int d = 1; d < 16; d <<= 1) ls += __shfl_xor(ls, d, 32);
      l[r] = l[r] * corr + ls;
      m[r] = mnew;
#pragma unroll
      for (int n = 0; n < 4; ++n) acc[n][r] *= corr;
      const int row = r + 8 * half;
      pl[row * 32 + l16]      = (__bf16)p0;   // C layout -> LDS [row][col]
      pl[row * 32 + 16 + l16] = (__bf16)p1;
    }
    asm volatile("s_wait_dscnt 0" ::: "memory");  // wave-local store->load order

    // ---- re-read P in A-operand layout (16x32, K = 32 keys) ----
    ABf pa;
    pa.h[0] = *(const v8bf*)(pl + l16 * 32 + half * 8);
    pa.h[1] = *(const v8bf*)(pl + l16 * 32 + 16 + half * 8);

    // ---- O += P * V (four 16x16 output tiles over dh) ----
#pragma unroll
    for (int n = 0; n < 4; ++n) acc[n] = wmma_bf16(pa, vb[n], acc[n]);
  }

  // ---- normalize and store (bf16, feeds final GEMM A operand) ----
#pragma unroll
  for (int n = 0; n < 4; ++n) {
#pragma unroll
    for (int r = 0; r < 8; ++r) {
      const int row = t0 + r + 8 * half;
      float v = acc[n][r] / l[r];
      O[(size_t)row * DMODEL + h * DHEAD + n * 16 + l16] = (__bf16)v;
    }
  }
}

// ------------------------------ launcher -----------------------------------

extern "C" void kernel_launch(void* const* d_in, const int* in_sizes, int n_in,
                              void* d_out, int out_size, void* d_ws, size_t ws_size,
                              hipStream_t stream) {
  (void)in_sizes; (void)n_in; (void)out_size; (void)ws_size;
  const float* query = (const float*)d_in[0];
  const float* key_  = (const float*)d_in[1];
  const float* value = (const float*)d_in[2];
  const float* Wq = (const float*)d_in[3];
  const float* bq = (const float*)d_in[4];
  const float* Wk = (const float*)d_in[5];
  const float* bk = (const float*)d_in[6];
  const float* Wv = (const float*)d_in[7];
  const float* bv = (const float*)d_in[8];
  const float* Wo = (const float*)d_in[9];
  const float* bo = (const float*)d_in[10];

  const size_t TD = (size_t)T_SEQ * DMODEL;   // 2M elems
  const size_t WW = (size_t)DMODEL * DMODEL;  // 1M elems
  char* ws = (char*)d_ws;
  size_t off = 0;
  auto alloc = [&](size_t bytes) -> char* {
    char* p = ws + off;
    off += (bytes + 255) & ~(size_t)255;
    return p;
  };
  __bf16* xq   = (__bf16*)alloc(TD * 2);   // bf16 query
  __bf16* xk   = (__bf16*)alloc(TD * 2);   // bf16 key
  __bf16* xv   = (__bf16*)alloc(TD * 2);   // bf16 value
  __bf16* WqT  = (__bf16*)alloc(WW * 2);
  __bf16* WkT  = (__bf16*)alloc(WW * 2);
  __bf16* WvT  = (__bf16*)alloc(WW * 2);
  __bf16* WoT  = (__bf16*)alloc(WW * 2);
  __bf16* Qr   = (__bf16*)alloc(TD * 2);   // rope'd Q  [T][1024]
  __bf16* Kr   = (__bf16*)alloc(TD * 2);   // rope'd K  [T][1024]
  __bf16* Vt   = (__bf16*)alloc(TD * 2);   // V^T       [1024][T]
  __bf16* attn = (__bf16*)alloc(TD * 2);   // attention output [T][1024]

  // 1) fp32 -> bf16 activations
  cvt_f32_bf16<<<(int)(TD / 256), 256, 0, stream>>>(query, xq, (int)TD);
  cvt_f32_bf16<<<(int)(TD / 256), 256, 0, stream>>>(key_,  xk, (int)TD);
  cvt_f32_bf16<<<(int)(TD / 256), 256, 0, stream>>>(value, xv, (int)TD);

  // 2) transpose weights to [N][K] bf16 (B-operand friendly)
  dim3 tg(DMODEL / 256, DMODEL);
  transpose_bf16<<<tg, 256, 0, stream>>>(Wq, WqT, DMODEL, DMODEL);
  transpose_bf16<<<tg, 256, 0, stream>>>(Wk, WkT, DMODEL, DMODEL);
  transpose_bf16<<<tg, 256, 0, stream>>>(Wv, WvT, DMODEL, DMODEL);
  transpose_bf16<<<tg, 256, 0, stream>>>(Wo, WoT, DMODEL, DMODEL);

  // 3) projections (+bias, +rope for Q/K, V transposed)
  gemm_wmma<<<256, 256, 0, stream>>>(xq, WqT, bq, Qr, 0);
  gemm_wmma<<<256, 256, 0, stream>>>(xk, WkT, bk, Kr, 0);
  gemm_wmma<<<256, 256, 0, stream>>>(xv, WvT, bv, Vt, 1);

  // 4) flash attention
  flash_attn<<<256, 256, 0, stream>>>(Qr, Kr, Vt, attn);

  // 5) output projection -> fp32 d_out
  gemm_wmma<<<256, 256, 0, stream>>>(attn, WoT, bo, d_out, 2);
}
